// SimpleVQVAE_33938831572995
// MI455X (gfx1250) — matrix operand order repaired and checked by reading, not compiled, not run on hardware
//
#include <hip/hip_runtime.h>
#include <hip/hip_bf16.h>

// ---------------------------------------------------------------------------
// Types
// ---------------------------------------------------------------------------
typedef __bf16 bf16_t;
typedef bf16_t v16bf __attribute__((ext_vector_type(16)));
typedef float  v8f   __attribute__((ext_vector_type(8)));
typedef float  v4f   __attribute__((ext_vector_type(4)));
typedef unsigned int v4u  __attribute__((ext_vector_type(4)));
typedef unsigned int v8u  __attribute__((ext_vector_type(8)));

union Frag16 {
    v16bf v;
    v4u   q[2];
    unsigned int u[8];
};

__device__ __forceinline__ unsigned short f2bf(float f) {
    unsigned int u = __float_as_uint(f);
    unsigned int r = u + 0x7FFFu + ((u >> 16) & 1u);   // round-to-nearest-even
    return (unsigned short)(r >> 16);
}

// LDS byte offset of a generic pointer to __shared__ (flat LDS aperture keeps
// the workgroup-relative offset in addr[31:0] per CDNA5 ISA 10.2).
__device__ __forceinline__ unsigned lds_offset(const void* p) {
    return (unsigned)(unsigned long long)(uintptr_t)p;
}

// ---------------------------------------------------------------------------
// Tensor Data Mover: async copy of a 2D bf16 tile [rows=128 x cols=32] with
// row stride K (elements) from global into LDS (packed 128x32, 8KB).
// D# built per CDNA5 ISA ch.8; issued once per block by wave 0.
// ---------------------------------------------------------------------------
__device__ __forceinline__ void tdm_load_tile_a(unsigned ldsOff,
                                                const unsigned short* gbase,
                                                int M, int K, int m0, int kt)
{
    unsigned long long ga = (unsigned long long)(uintptr_t)gbase +
                            (((unsigned long long)(unsigned)m0 * (unsigned)K +
                              (unsigned)kt) << 1);
    v4u g0;
    g0[0] = 1u;                                   // count=1, user descriptor
    g0[1] = ldsOff;                               // lds_addr (bytes)
    g0[2] = (unsigned)ga;                         // global_addr[31:0]
    g0[3] = (unsigned)((ga >> 32) & 0x1FFFFFFu)   // global_addr[56:32]
            | 0x80000000u;                        // type=2 ("image")
    v8u g1;
    g1[0] = 1u << 16;                                             // data_size=2B
    g1[1] = ((unsigned)K & 0xFFFFu) << 16;                        // tensor_dim0 lo
    g1[2] = (((unsigned)K >> 16) & 0xFFFFu)                       // tensor_dim0 hi
            | (((unsigned)M & 0xFFFFu) << 16);                    // tensor_dim1 lo
    g1[3] = (((unsigned)M >> 16) & 0xFFFFu) | (32u << 16);        // dim1 hi | tile_dim0=32
    g1[4] = 128u;                                                 // tile_dim1=128, tile_dim2=0
    g1[5] = (unsigned)K;                                          // tensor_dim0_stride lo
    g1[6] = 0u;                                                   // stride0 hi | stride1 lo
    g1[7] = 0u;                                                   // stride1 hi
    v4u gz = {0u, 0u, 0u, 0u};                                    // groups 2/3 unused
    asm volatile("tensor_load_to_lds %0, %1, %2, %3"
                 :: "s"(g0), "s"(g1), "s"(gz), "s"(gz)
                 : "memory");
}

// ---------------------------------------------------------------------------
// f32 -> bf16 bulk conversion (n multiple of 4)
// ---------------------------------------------------------------------------
__global__ __launch_bounds__(256)
void cvt_bf16_kernel(const float* __restrict__ in, unsigned short* __restrict__ out, int n)
{
    const int i = (blockIdx.x * 256 + threadIdx.x) * 4;
    if (i + 3 < n) {
        v4f f = *(const v4f*)(in + i);
        unsigned p0 = (unsigned)f2bf(f[0]) | ((unsigned)f2bf(f[1]) << 16);
        unsigned p1 = (unsigned)f2bf(f[2]) | ((unsigned)f2bf(f[3]) << 16);
        *(unsigned*)&out[i]     = p0;
        *(unsigned*)&out[i + 2] = p1;
    }
}

// ---------------------------------------------------------------------------
// Fused GEMM: C = act(A @ W + bias)
//   A: [M,K] row-major; bf16 (A_TDM path, async tensor loads) or f32 (layer 1)
//   W: [K,N] row-major bf16 (pre-converted)
//   C: [M,N] f32 or bf16
// Block tile 128x128, K-step 32, 8 waves (wave32), wave tile 32x64 via
// 8x v_wmma_f32_16x16x32_bf16. Double-buffered LDS, one barrier per K-step.
// ---------------------------------------------------------------------------
template <bool A_TDM, bool OUT_F32, bool RELU, bool CLIP>
__global__ __launch_bounds__(256)
void gemm_kernel(const void* __restrict__ Aptr, const unsigned short* __restrict__ W,
                 const float* __restrict__ bias, void* __restrict__ Cptr,
                 int M, int N, int K)
{
    constexpr int BM = 128, BN = 128, BK = 32;
    __shared__ __align__(16) unsigned short sA[2][BM][BK];   // [m][k]
    __shared__ __align__(16) unsigned short sB[2][BN][BK];   // [n][k] transposed

    const int tid  = threadIdx.x;
    const int lane = tid & 31;
    const int wave = tid >> 5;
    const int m0 = blockIdx.x * BM;
    const int n0 = blockIdx.y * BN;
    const int wm = (wave >> 1) * 32;
    const int wn = (wave & 1) * 64;

    v8f acc[2][4];
#pragma unroll
    for (int i = 0; i < 2; ++i)
#pragma unroll
        for (int j = 0; j < 4; ++j)
#pragma unroll
            for (int r = 0; r < 8; ++r) acc[i][j][r] = 0.0f;

    const int l15  = lane & 15;
    const int kh8  = (lane >> 4) * 8;
    const int kh16 = (lane >> 4) * 16;

    // loader thread mapping
    const int arow = tid >> 1, akq = (tid & 1) * 16;         // A (f32 path)
    const int bk = tid >> 3, bnq = (tid & 7) * 16;           // B

    const unsigned ldsA0 = lds_offset(&sA[0][0][0]);
    const int nk = K / BK;

    // ---- prologue: prefetch tile 0 ----
    v8u bReg;                      // 16 bf16 of B
    unsigned aReg[8];              // 16 bf16 of A (f32 path only)
    {
        const unsigned short* bs = W + (size_t)bk * N + n0 + bnq;
        bReg.lo = *(const v4u*)bs;
        bReg.hi = *(const v4u*)(bs + 8);
        if (A_TDM) {
            if (wave == 0)
                tdm_load_tile_a(ldsA0, (const unsigned short*)Aptr, M, K, m0, 0);
        } else {
            const float* as = (const float*)Aptr + (size_t)(m0 + arow) * K + akq;
#pragma unroll
            for (int i = 0; i < 4; ++i) {
                v4f f = *(const v4f*)(as + 4 * i);
                aReg[2 * i]     = (unsigned)f2bf(f[0]) | ((unsigned)f2bf(f[1]) << 16);
                aReg[2 * i + 1] = (unsigned)f2bf(f[2]) | ((unsigned)f2bf(f[3]) << 16);
            }
        }
    }

    for (int it = 0; it < nk; ++it) {
        const int cur = it & 1;
        // ---- store staged tile(it) to LDS ----
        if (!A_TDM) {
            *(v4u*)&sA[cur][arow][akq]     = (v4u){aReg[0], aReg[1], aReg[2], aReg[3]};
            *(v4u*)&sA[cur][arow][akq + 8] = (v4u){aReg[4], aReg[5], aReg[6], aReg[7]};
        }
#pragma unroll
        for (int i = 0; i < 4; ++i) {
            unsigned u0 = bReg[i], u1 = bReg[4 + i];
            sB[cur][bnq + 2 * i][bk]     = (unsigned short)(u0 & 0xFFFFu);
            sB[cur][bnq + 2 * i + 1][bk] = (unsigned short)(u0 >> 16);
            sB[cur][bnq + 8 + 2 * i][bk]     = (unsigned short)(u1 & 0xFFFFu);
            sB[cur][bnq + 8 + 2 * i + 1][bk] = (unsigned short)(u1 >> 16);
        }
        if (A_TDM && wave == 0) __builtin_amdgcn_s_wait_tensorcnt(0);
        __syncthreads();   // tile(it) visible; all reads of buffer (it-1) done

        // ---- prefetch tile(it+1) ----
        if (it + 1 < nk) {
            const int kt = (it + 1) * BK;
            const unsigned short* bs = W + (size_t)(kt + bk) * N + n0 + bnq;
            bReg.lo = *(const v4u*)bs;
            bReg.hi = *(const v4u*)(bs + 8);
            if (A_TDM) {
                if (wave == 0)
                    tdm_load_tile_a(ldsA0 + (unsigned)((it + 1) & 1) * (BM * BK * 2),
                                    (const unsigned short*)Aptr, M, K, m0, kt);
            } else {
                const float* as = (const float*)Aptr + (size_t)(m0 + arow) * K + kt + akq;
#pragma unroll
                for (int i = 0; i < 4; ++i) {
                    v4f f = *(const v4f*)(as + 4 * i);
                    aReg[2 * i]     = (unsigned)f2bf(f[0]) | ((unsigned)f2bf(f[1]) << 16);
                    aReg[2 * i + 1] = (unsigned)f2bf(f[2]) | ((unsigned)f2bf(f[3]) << 16);
                }
            }
        }

        // ---- fragments (ISA 16-bit layouts) + 8 WMMA ----
        Frag16 a0, a1, bf[4];
        a0.q[0] = *(const v4u*)&sA[cur][wm + l15][kh8];
        a0.q[1] = *(const v4u*)&sA[cur][wm + l15][16 + kh8];
        a1.q[0] = *(const v4u*)&sA[cur][wm + 16 + l15][kh8];
        a1.q[1] = *(const v4u*)&sA[cur][wm + 16 + l15][16 + kh8];
#pragma unroll
        for (int j = 0; j < 4; ++j) {
            const int n = wn + j * 16 + l15;
            bf[j].q[0] = *(const v4u*)&sB[cur][n][kh8];
            bf[j].q[1] = *(const v4u*)&sB[cur][n][16 + kh8];
        }
#pragma unroll
        for (int j = 0; j < 4; ++j) {
            acc[0][j] = __builtin_amdgcn_wmma_f32_16x16x32_bf16(
                false, a0.v, false, bf[j].v, (short)0, acc[0][j], false, false);
            acc[1][j] = __builtin_amdgcn_wmma_f32_16x16x32_bf16(
                false, a1.v, false, bf[j].v, (short)0, acc[1][j], false, false);
        }
    }

    // ---- epilogue: bias + activation + store ----
    const int kh = lane >> 4;
#pragma unroll
    for (int i = 0; i < 2; ++i) {
#pragma unroll
        for (int j = 0; j < 4; ++j) {
            const int n = n0 + wn + j * 16 + l15;
            const float bv = bias[n];
#pragma unroll
            for (int r = 0; r < 8; ++r) {
                const int m = m0 + wm + i * 16 + r + 8 * kh;
                float v = acc[i][j][r] + bv;
                if (RELU) v = fmaxf(v, 0.0f);
                if (CLIP) v = fminf(fmaxf(v, -1.0f), 1.0f);
                if (OUT_F32)
                    ((float*)Cptr)[(size_t)m * N + n] = v;
                else
                    ((unsigned short*)Cptr)[(size_t)m * N + n] = f2bf(v);
            }
        }
    }
}

// ---------------------------------------------------------------------------
// VQ per-expert kernel: grid (128, E). One wave per row.
// ---------------------------------------------------------------------------
__global__ __launch_bounds__(256)
void vq_expert_kernel(const float* __restrict__ latent, const float* __restrict__ Win,
                      const float* __restrict__ bin_, const float* __restrict__ codebook,
                      const float* __restrict__ Wout, const float* __restrict__ bout,
                      float* __restrict__ qemb, float* __restrict__ idx_out,
                      float* __restrict__ accum)
{
    const int e = blockIdx.y;
    __shared__ float s_cb[256][33];
    __shared__ float s_lat[8][128];
    __shared__ float s_xe[8][32];
    __shared__ float s_q[8][32];

    const int tid = threadIdx.x, lane = tid & 31, wave = tid >> 5;

    for (int i = tid; i < 256 * 32; i += 256)
        s_cb[i >> 5][i & 31] = codebook[e * 8192 + i];
    __syncthreads();

    const float* WinE  = Win  + e * 128 * 32;
    const float* WoutE = Wout + e * 32 * 128;
    const float  binc  = bin_[e * 32 + lane];
    float commitAcc = 0.0f;

    for (int it = 0; it < 8; ++it) {
        const int row = blockIdx.x * 64 + it * 8 + wave;
#pragma unroll
        for (int j = 0; j < 4; ++j)
            s_lat[wave][lane + 32 * j] = latent[(size_t)row * 128 + lane + 32 * j];
        __syncthreads();

        float xe = binc;
#pragma unroll 8
        for (int h = 0; h < 128; ++h) xe += s_lat[wave][h] * WinE[h * 32 + lane];
        s_xe[wave][lane] = xe;
        __syncthreads();

        float bestD = 3.4e38f; int bestI = 0;
#pragma unroll
        for (int j = 0; j < 8; ++j) {
            const int n = lane + 32 * j;
            float d = 0.0f;
#pragma unroll
            for (int c = 0; c < 32; ++c) {
                const float t = s_xe[wave][c] - s_cb[n][c];
                d += t * t;
            }
            if (d < bestD || (d == bestD && n < bestI)) { bestD = d; bestI = n; }
        }
#pragma unroll
        for (int off = 16; off; off >>= 1) {
            const float od = __shfl_xor(bestD, off, 32);
            const int   oi = __shfl_xor(bestI, off, 32);
            if (od < bestD || (od == bestD && oi < bestI)) { bestD = od; bestI = oi; }
        }

        const float qc = s_cb[bestI][lane];
        s_q[wave][lane] = qc;
        float cs = (qc - xe) * (qc - xe);
#pragma unroll
        for (int off = 16; off; off >>= 1) cs += __shfl_xor(cs, off, 32);
        commitAcc += cs;
        __syncthreads();

#pragma unroll
        for (int j = 0; j < 4; ++j) {
            const int h = lane + 32 * j;
            float v = bout[e * 128 + h];
#pragma unroll
            for (int c = 0; c < 32; ++c) v += s_q[wave][c] * WoutE[c * 128 + h];
            qemb[((size_t)e * 8192 + row) * 128 + h] = v;
        }
        if (lane == 0) idx_out[row * 3 + e] = (float)bestI;
        __syncthreads();
    }
    if (lane == 0) atomicAdd(&accum[0], commitAcc);
}

// ---------------------------------------------------------------------------
// Reduce: rl(bf16) = mean_e(qemb); accumulate M[e][h] and 6 second moments.
// accum: [0]=commit, [1..6]=S(00,11,22,01,02,12), [16..400)=M
// ---------------------------------------------------------------------------
__global__ __launch_bounds__(256)
void vq_reduce_kernel(const float* __restrict__ qemb, unsigned short* __restrict__ rl,
                      float* __restrict__ accum)
{
    __shared__ float s_M[8][384];
    const int tid = threadIdx.x, lane = tid & 31, wave = tid >> 5;
    for (int i = tid; i < 8 * 384; i += 256) (&s_M[0][0])[i] = 0.0f;
    __syncthreads();

    float pair[6] = {0, 0, 0, 0, 0, 0};
    for (int it = 0; it < 8; ++it) {
        const int row = blockIdx.x * 64 + it * 8 + wave;
        float q0[4], q1[4], q2[4];
#pragma unroll
        for (int j = 0; j < 4; ++j) {
            const int h = lane + 32 * j;
            q0[j] = qemb[((size_t)0 * 8192 + row) * 128 + h];
            q1[j] = qemb[((size_t)1 * 8192 + row) * 128 + h];
            q2[j] = qemb[((size_t)2 * 8192 + row) * 128 + h];
            rl[(size_t)row * 128 + h] = f2bf((q0[j] + q1[j] + q2[j]) * (1.0f / 3.0f));
            s_M[wave][0 * 128 + h] += q0[j];
            s_M[wave][1 * 128 + h] += q1[j];
            s_M[wave][2 * 128 + h] += q2[j];
        }
#pragma unroll
        for (int j = 0; j < 4; ++j) {
            pair[0] += q0[j] * q0[j];
            pair[1] += q1[j] * q1[j];
            pair[2] += q2[j] * q2[j];
            pair[3] += q0[j] * q1[j];
            pair[4] += q0[j] * q2[j];
            pair[5] += q1[j] * q2[j];
        }
    }
#pragma unroll
    for (int k = 0; k < 6; ++k) {
        float v = pair[k];
#pragma unroll
        for (int off = 16; off; off >>= 1) v += __shfl_xor(v, off, 32);
        pair[k] = v;
    }
    if (lane == 0)
        for (int k = 0; k < 6; ++k) atomicAdd(&accum[1 + k], pair[k]);
    __syncthreads();
    for (int i = tid; i < 384; i += 256) {
        float s = 0.0f;
        for (int w = 0; w < 8; ++w) s += s_M[w][i];
        atomicAdd(&accum[16 + i], s);
    }
}

// ---------------------------------------------------------------------------
__global__ void zero_accum_kernel(float* accum)
{
    const int i = threadIdx.x;
    if (i < 512) accum[i] = 0.0f;
}

__global__ void vq_finalize_kernel(const float* __restrict__ accum, float* __restrict__ outs)
{
    if (threadIdx.x != 0 || blockIdx.x != 0) return;
    const float Bf = 8192.0f, Hf = 128.0f;
    float S[3][3];
    S[0][0] = accum[1]; S[1][1] = accum[2]; S[2][2] = accum[3];
    S[0][1] = S[1][0] = accum[4];
    S[0][2] = S[2][0] = accum[5];
    S[1][2] = S[2][1] = accum[6];
    float cov[3][3];
    for (int e = 0; e < 3; ++e)
        for (int f = 0; f < 3; ++f) {
            float cross = 0.0f;
            for (int h = 0; h < 128; ++h)
                cross += accum[16 + e * 128 + h] * accum[16 + f * 128 + h];
            cross /= Bf;
            cov[e][f] = (S[e][f] - cross) / (Bf * Hf - 1.0f);
        }
    float sd[3];
    for (int e = 0; e < 3; ++e) {
        const float s = sqrtf(cov[e][e]);
        sd[e] = (s > 1e-8f) ? s : 1.0f;
    }
    float dec = 0.0f;
    for (int e = 0; e < 3; ++e)
        for (int f = 0; f < 3; ++f) {
            if (e == f) continue;
            const float c = cov[e][f] / (sd[e] * sd[f]);
            dec += c * c;
        }
    outs[0] = 0.25f * accum[0] / (Bf * 32.0f);
    outs[1] = dec;
}

// ---------------------------------------------------------------------------
extern "C" void kernel_launch(void* const* d_in, const int* in_sizes, int n_in,
                              void* d_out, int out_size, void* d_ws, size_t ws_size,
                              hipStream_t stream)
{
    const float* x    = (const float*)d_in[0];
    const float* Wd1  = (const float*)d_in[1];
    const float* bd1  = (const float*)d_in[2];
    const float* Wd2  = (const float*)d_in[3];
    const float* bd2  = (const float*)d_in[4];
    const float* Wd3  = (const float*)d_in[5];
    const float* bd3  = (const float*)d_in[6];
    const float* Win  = (const float*)d_in[7];
    const float* bin_ = (const float*)d_in[8];
    const float* cb   = (const float*)d_in[9];
    const float* Wout = (const float*)d_in[10];
    const float* bout = (const float*)d_in[11];
    const float* Wu1  = (const float*)d_in[12];
    const float* bu1  = (const float*)d_in[13];
    const float* Wu2  = (const float*)d_in[14];
    const float* bu2  = (const float*)d_in[15];
    const float* Wu3  = (const float*)d_in[16];
    const float* bu3  = (const float*)d_in[17];

    const size_t MB = 1024ull * 1024ull;
    char* ws = (char*)d_ws;
    // Region [0,12MB): h1(0..8M)+h2(8..12M) -> qemb(0..12M) -> u1(0..4M)+u2(4..12M)
    unsigned short* h1   = (unsigned short*)(ws);
    unsigned short* h2   = (unsigned short*)(ws + 8 * MB);
    float*          qemb = (float*)(ws);
    unsigned short* u1   = (unsigned short*)(ws);
    unsigned short* u2   = (unsigned short*)(ws + 4 * MB);
    float*          lat  = (float*)(ws + 12 * MB);
    unsigned short* rl   = (unsigned short*)(ws + 16 * MB);
    float*          acc  = (float*)(ws + 18 * MB);
    // bf16 weights at [19MB, ~28MB)
    unsigned short* wd1b = (unsigned short*)(ws + 19 * MB);
    unsigned short* wd2b = wd1b + 2097152;
    unsigned short* wd3b = wd2b + 131072;
    unsigned short* wu1b = wd3b + 32768;
    unsigned short* wu2b = wu1b + 32768;
    unsigned short* wu3b = wu2b + 131072;

    float* recon   = (float*)d_out;
    float* idx_out = recon + (size_t)8192 * 4096;
    float* scalars = idx_out + (size_t)8192 * 3;

    zero_accum_kernel<<<1, 512, 0, stream>>>(acc);

    // one-time (per call) weight conversion to bf16
    cvt_bf16_kernel<<<2048, 256, 0, stream>>>(Wd1, wd1b, 2097152);
    cvt_bf16_kernel<<<128,  256, 0, stream>>>(Wd2, wd2b, 131072);
    cvt_bf16_kernel<<<32,   256, 0, stream>>>(Wd3, wd3b, 32768);
    cvt_bf16_kernel<<<32,   256, 0, stream>>>(Wu1, wu1b, 32768);
    cvt_bf16_kernel<<<128,  256, 0, stream>>>(Wu2, wu2b, 131072);
    cvt_bf16_kernel<<<2048, 256, 0, stream>>>(Wu3, wu3b, 2097152);

    // down MLP: 4096 -> 512 -> 256 -> 128
    gemm_kernel<false, false, true,  false><<<dim3(64, 4), 256, 0, stream>>>(x,  wd1b, bd1, h1,  8192, 512,  4096);
    gemm_kernel<true,  false, true,  false><<<dim3(64, 2), 256, 0, stream>>>(h1, wd2b, bd2, h2,  8192, 256,  512);
    gemm_kernel<true,  true,  false, false><<<dim3(64, 1), 256, 0, stream>>>(h2, wd3b, bd3, lat, 8192, 128,  256);

    // VQ experts + reduction
    vq_expert_kernel<<<dim3(128, 3), 256, 0, stream>>>(lat, Win, bin_, cb, Wout, bout, qemb, idx_out, acc);
    vq_reduce_kernel<<<128, 256, 0, stream>>>(qemb, rl, acc);

    // up MLP: 128 -> 256 -> 512 -> 4096 (+clip)
    gemm_kernel<true, false, true,  false><<<dim3(64, 2),  256, 0, stream>>>(rl, wu1b, bu1, u1,    8192, 256,  128);
    gemm_kernel<true, false, true,  false><<<dim3(64, 4),  256, 0, stream>>>(u1, wu2b, bu2, u2,    8192, 512,  256);
    gemm_kernel<true, true,  false, true ><<<dim3(64, 32), 256, 0, stream>>>(u2, wu3b, bu3, recon, 8192, 4096, 512);

    vq_finalize_kernel<<<1, 1, 0, stream>>>(acc, scalars);
}
